// BinaryClassifier_20160576487456
// MI455X (gfx1250) — matrix operand order repaired and checked by reading, not compiled, not run on hardware
//
#include <hip/hip_runtime.h>

// ---------------------------------------------------------------------------
// Packed-LSTM (B=64, S=2048, H=512, V=25) + linear head on MI455X (gfx1250).
//
//  * 4 batch groups x 16 batch (N=16 == WMMA N), independent (no cross sync).
//  * 16 WGs per group own 128 gate rows each (32 hidden units x 4 gates).
//  * W_hh slice lives in REGISTERS as fp16 WMMA A-fragments (16 x v16h =
//    128 VGPRs/lane; 2 waves/SIMD so this fits), gathered once from global.
//  * Per step/wave: 16 x v_wmma_f32_16x16x32_f16 (K=512), B fragments from
//    LDS; one-hot x@W_ih + biases folded into the WMMA C initializer.
//  * h (16KB/group) double-buffered in L2; staged into LDS each step with
//    GLOBAL_LOAD_ASYNC_TO_LDS_B128 + s_wait_asynccnt (CDNA5 async path).
//  * One monotonic L2-atomic barrier per step among the 16 WGs of a group.
// ---------------------------------------------------------------------------

typedef __attribute__((ext_vector_type(16))) _Float16 v16h;
typedef __attribute__((ext_vector_type(8)))  float    v8f;

#define VOCAB   25
#define HID     512
#define SEQ     2048
#define BATCH   64
#define NB      16      // batch per group (== WMMA N)
#define NGROUP  4
#define NROWWG  16      // WGs per group
#define ROWS    128     // gate rows per WG (32 hidden x 4 gates)
#define HSTR    528     // padded LDS stride for h (halves); 1056B = 33*32B

struct __align__(64) Smem {
    _Float16 h[NB * HSTR];          // 16896B : h[n][k] fp16, padded
    float    gates[ROWS * NB];      // 8192B  : gate pre-activations
    float    wihb[ROWS * VOCAB];    // 12800B : W_ih slice + (b_ih + b_hh) folded in
    float    c[32 * NB];            // 2048B  : cell state (WG-private)
    int      ord[NB];
    int      len[NB];
    int      tok[NB];
};

__device__ __forceinline__ float sigm(float x) {
    return 1.0f / (1.0f + __expf(-x));
}

// ---------------------------------------------------------------------------
// Prologue: stable descending argsort of lengths (== jnp.argsort(-lengths)),
// zero the double-buffered h and the barrier counters.  ws layout:
//   [0,256)    int   order[64]
//   [256,512)  int   lens_sorted[64]
//   [512,768)  uint  bar[4*16]   (one cacheline per group)
//   [1024, +131072) _Float16 hbuf[2][4][16][512]
// ---------------------------------------------------------------------------
__global__ void lstm_prologue(const int* __restrict__ lengths, char* __restrict__ ws) {
    __shared__ int len_s[BATCH];
    const int tid = threadIdx.x;
    int*      order = (int*)(ws);
    int*      lsort = (int*)(ws + 256);
    unsigned* bar   = (unsigned*)(ws + 512);
    _Float16* hbuf  = (_Float16*)(ws + 1024);

    if (tid < BATCH) len_s[tid] = lengths[tid];
    __syncthreads();
    if (tid < BATCH) {
        const int li = len_s[tid];
        int rank = 0;
        #pragma unroll 8
        for (int j = 0; j < BATCH; ++j) {
            const int lj = len_s[j];
            if (lj > li || (lj == li && j < tid)) rank++;
        }
        order[rank] = tid;
        lsort[rank] = li;
    }
    if (tid < 64) bar[tid] = 0u;
    for (int i = tid; i < 2 * NGROUP * NB * HID; i += 256)
        hbuf[i] = (_Float16)0.0f;
}

// ---------------------------------------------------------------------------
// Persistent LSTM kernel: 64 WGs x 256 threads (8 wave32), 1 WG / WGP.
// ---------------------------------------------------------------------------
__global__ __launch_bounds__(256, 1) void lstm_persistent(
    const int*   __restrict__ tokens,  // [64][2048]
    const float* __restrict__ W_ih,    // [2048][25]
    const float* __restrict__ W_hh,    // [2048][512]
    const float* __restrict__ b_ih,    // [2048]
    const float* __restrict__ b_hh,    // [2048]
    const float* __restrict__ fc_w,    // [512]
    const float* __restrict__ fc_b,    // [1]
    float*       __restrict__ out,     // [64]
    char*        __restrict__ ws) {

    __shared__ Smem sm;

    const int tid  = threadIdx.x;
    const int lane = tid & 31;
    const int wv   = tid >> 5;            // wave id 0..7 -> 16-row output tile
    const int g    = blockIdx.x >> 4;     // batch group 0..3
    const int r    = blockIdx.x & 15;     // row-chunk (hidden units r*32..r*32+31)

    int*      order = (int*)(ws);
    int*      lsort = (int*)(ws + 256);
    unsigned* bar   = (unsigned*)(ws + 512);
    _Float16* hbuf  = (_Float16*)(ws + 1024);   // [2][4][16][512]
    const int HB_G  = NB * HID;
    const int HB_B  = NGROUP * HB_G;

    // ---- one-time setup -------------------------------------------------
    if (tid < NB) {
        sm.ord[tid] = order[g * NB + tid];
        sm.len[tid] = lsort[g * NB + tid];
    }
    // One-hot gather table with biases pre-folded:
    // local row lr: gate q = lr/32, hidden j = r*32 + lr%32, global row = q*512 + j.
    for (int i = tid; i < ROWS * VOCAB; i += 256) {
        const int lr = i / VOCAB, vv = i % VOCAB;
        const int grow = (lr >> 5) * HID + r * 32 + (lr & 31);
        sm.wihb[i] = W_ih[grow * VOCAB + vv] + b_ih[grow] + b_hh[grow];
    }
    for (int i = tid; i < 32 * NB; i += 256) sm.c[i] = 0.0f;

    // W_hh slice -> fp16 WMMA A-fragments, register-resident (one-time global
    // gather).  A k order used here (B matches): M = lane%16; K = s for
    // lanes 0-15, +16 for lanes 16-31.  Per lane the row is fixed.
    v16h wreg[16];
    {
        const int lr   = wv * 16 + (lane & 15);
        const int grow = (lr >> 5) * HID + r * 32 + (lr & 31);
        const float* wrow  = W_hh + grow * HID;
        const int    koffb = (lane >= 16) ? 16 : 0;
        #pragma unroll
        for (int kb = 0; kb < 16; ++kb) {
            #pragma unroll
            for (int s = 0; s < 16; ++s) {
                const int k = kb * 32 + s + koffb;
                wreg[kb][s] = (_Float16)wrow[k];
            }
        }
    }
    __syncthreads();

    const int nn = lane & 15;                         // C/D + B: N = lane%16
    const int hi = lane >> 4;                         // C/D: M = v + 8*hi
    const _Float16* hbase = sm.h + nn * HSTR + hi * 16;

    // ---- time loop ------------------------------------------------------
    unsigned bar_target = 0;
    for (int t = 0; t < SEQ; ++t) {
        // Stage h (read buffer) into LDS via CDNA5 async global->LDS copies:
        // 64B per thread = 4 x GLOBAL_LOAD_ASYNC_TO_LDS_B128 (ASYNCcnt).
        {
            const _Float16* hread = hbuf + (size_t)(t & 1) * HB_B + g * HB_G;
            const int n = tid >> 4, ch = tid & 15;
            const unsigned lds0 =
                (unsigned)(uintptr_t)(sm.h + n * HSTR + ch * 32);
            const unsigned long long ga =
                (unsigned long long)(uintptr_t)(hread + n * HID + ch * 32);
            #pragma unroll
            for (int q = 0; q < 4; ++q) {
                asm volatile("global_load_async_to_lds_b128 %0, %1, off"
                             :: "v"(lds0 + (unsigned)(q * 16)),
                                "v"(ga + (unsigned long long)(q * 16))
                             : "memory");
            }
        }
        if (tid < NB) sm.tok[tid] = tokens[sm.ord[tid] * SEQ + t];
        asm volatile("s_wait_asynccnt 0x0" ::: "memory");
        __syncthreads();

        // WMMA: wave wv computes gate tile rows [wv*16, wv*16+16) x 16 batch.
        {
            const int tk = sm.tok[nn];
            v8f acc;
            #pragma unroll
            for (int v = 0; v < 8; ++v) {
                const int lr2 = wv * 16 + v + hi * 8;
                acc[v] = sm.wihb[lr2 * VOCAB + tk];   // one-hot x@W_ih + biases
            }
            #pragma unroll
            for (int kb = 0; kb < 16; ++kb) {
                v16h b = *(const v16h*)(hbase + kb * 32);
                acc = __builtin_amdgcn_wmma_f32_16x16x32_f16(
                    false, wreg[kb], false, b, (short)0, acc, false, false);
            }
            #pragma unroll
            for (int v = 0; v < 8; ++v) {
                sm.gates[(wv * 16 + v + hi * 8) * NB + nn] = acc[v];
            }
        }
        __syncthreads();

        // Elementwise LSTM cell update for this WG's 32 hidden units.
        {
            _Float16* hwrite = hbuf + (size_t)((t + 1) & 1) * HB_B + g * HB_G;
            #pragma unroll
            for (int rep = 0; rep < 2; ++rep) {
                const int item = tid + rep * 256;        // 512 = 32 j x 16 n
                const int j = item >> 4, n = item & 15;
                const float iv = sigm(sm.gates[(     j) * NB + n]);
                const float fv = sigm(sm.gates[(32 + j) * NB + n]);
                const float gv = tanhf(sm.gates[(64 + j) * NB + n]);
                const float ov = sigm(sm.gates[(96 + j) * NB + n]);
                const float cold = sm.c[j * NB + n];
                const float cnew = fv * cold + iv * gv;
                const float hnew = ov * tanhf(cnew);
                const int   kg   = r * 32 + j;
                const float hold = (float)sm.h[n * HSTR + kg];
                const bool  act  = (t < sm.len[n]);      // freeze past seq end
                if (act) sm.c[j * NB + n] = cnew;
                hwrite[n * HID + kg] = (_Float16)(act ? hnew : hold);
            }
        }

        // One barrier per step across the 16 WGs of this group.
        __syncthreads();
        bar_target += NROWWG;
        if (tid == 0) {
            __threadfence();                         // release h stores
            atomicAdd(&bar[g * 16], 1u);
            volatile unsigned* p = &bar[g * 16];
            while (*p < bar_target) { __builtin_amdgcn_s_sleep(2); }
            __threadfence();                         // acquire peers' h stores
        }
        __syncthreads();
    }

    // ---- head: out = h_last @ fc_w + fc_b (deterministic, r==0 WG only) --
    if (r == 0 && tid < NB) {
        const _Float16* hf = hbuf + (size_t)(SEQ & 1) * HB_B + g * HB_G + tid * HID;
        float acc = fc_b[0];
        #pragma unroll 8
        for (int k = 0; k < HID; ++k) acc += (float)hf[k] * fc_w[k];
        out[g * NB + tid] = acc;
    }
}

// ---------------------------------------------------------------------------
extern "C" void kernel_launch(void* const* d_in, const int* in_sizes, int n_in,
                              void* d_out, int out_size, void* d_ws, size_t ws_size,
                              hipStream_t stream) {
    const int*   tokens = (const int*)  d_in[0];
    const int*   lens   = (const int*)  d_in[1];
    const float* W_ih   = (const float*)d_in[2];
    const float* W_hh   = (const float*)d_in[3];
    const float* b_ih   = (const float*)d_in[4];
    const float* b_hh   = (const float*)d_in[5];
    const float* fc_w   = (const float*)d_in[6];
    const float* fc_b   = (const float*)d_in[7];
    float*       out    = (float*)d_out;
    char*        ws     = (char*)d_ws;

    lstm_prologue<<<1, 256, 0, stream>>>(lens, ws);
    lstm_persistent<<<NGROUP * NROWWG, 256, 0, stream>>>(
        tokens, W_ih, W_hh, b_ih, b_hh, fc_w, fc_b, out, ws);
}
